// FusedProjectionChain_62302795596453
// MI455X (gfx1250) — matrix operand order repaired and checked by reading, not compiled, not run on hardware
//
#include <hip/hip_runtime.h>
#include <hip/hip_bf16.h>

typedef __attribute__((ext_vector_type(16))) __bf16 v16bf;
typedef __attribute__((ext_vector_type(8)))  __bf16 v8bf;
typedef __attribute__((ext_vector_type(8)))  float  v8f;
typedef __attribute__((ext_vector_type(4)))  float  v4f;

#define BM 256
#define BN 64
#define BK 64
#define LDK 72   // padded LDS row stride (bf16 elems); 72*2=144B, 16B aligned

#define M_TOTAL 32768L
#define N_TOTAL 4112
#define K_TOTAL 1024

__global__ __launch_bounds__(256)
void fused_proj_wmma_kernel(const float* __restrict__ x,
                            const float* __restrict__ W,
                            float* __restrict__ out)
{
    __shared__ __bf16 Alds[BM][LDK];   // 36 KB
    __shared__ __bf16 Blds[BN][LDK];   //  9 KB

    const int tid  = threadIdx.x;
    const int lane = tid & 31;
    const int wave = tid >> 5;        // 0..7 -> wave's 32-row strip
    const int half = lane >> 4;       // 0/1
    const int l16  = lane & 15;

    const long m0 = (long)blockIdx.y * BM;   // 128 m-blocks
    const int  n0 = blockIdx.x * BN;         // 65 n-blocks (last partial: alpha)

    // accumulators: 2 M-subtiles x 4 N-subtiles of 16x16
    v8f acc[2][4];
    #pragma unroll
    for (int i = 0; i < 2; ++i)
        #pragma unroll
        for (int j = 0; j < 4; ++j) acc[i][j] = (v8f){};

    // staging: 4 threads per row, 16 floats (4 x float4) each
    const int srow = tid >> 2;        // 0..63
    const int sk   = (tid & 3) * 16;  // 0,16,32,48

    const int  nW     = n0 + srow;
    const bool wvalid = (nW < N_TOTAL);
    const float* wsrc = W + (long)(wvalid ? nW : 0) * K_TOTAL;

    for (int kt = 0; kt < K_TOTAL; kt += BK) {
        // ---- stage A tile: 256 rows x 64 k, fp32 -> bf16, 4 passes ----
        #pragma unroll
        for (int p = 0; p < 4; ++p) {
            const int row = p * 64 + srow;
            const float* src = x + (m0 + row) * K_TOTAL + kt + sk;
            if (kt + BK < K_TOTAL) __builtin_prefetch(src + BK, 0, 1);
            v4f v[4];
            #pragma unroll
            for (int i = 0; i < 4; ++i) v[i] = *(const v4f*)(src + i * 4);
            v8bf c0, c1;
            #pragma unroll
            for (int j = 0; j < 4; ++j) { c0[j] = (__bf16)v[0][j]; c0[j + 4] = (__bf16)v[1][j]; }
            #pragma unroll
            for (int j = 0; j < 4; ++j) { c1[j] = (__bf16)v[2][j]; c1[j + 4] = (__bf16)v[3][j]; }
            *(v8bf*)(&Alds[row][sk])     = c0;
            *(v8bf*)(&Alds[row][sk + 8]) = c1;
        }
        // ---- stage B tile: 64 rows x 64 k (W rows, zero-pad OOB) ----
        {
            v4f v[4];
            #pragma unroll
            for (int i = 0; i < 4; ++i) {
                v[i] = *(const v4f*)(wsrc + kt + sk + i * 4);
                if (!wvalid) v[i] = (v4f){0.f, 0.f, 0.f, 0.f};
            }
            v8bf c0, c1;
            #pragma unroll
            for (int j = 0; j < 4; ++j) { c0[j] = (__bf16)v[0][j]; c0[j + 4] = (__bf16)v[1][j]; }
            #pragma unroll
            for (int j = 0; j < 4; ++j) { c1[j] = (__bf16)v[2][j]; c1[j + 4] = (__bf16)v[3][j]; }
            *(v8bf*)(&Blds[srow][sk])     = c0;
            *(v8bf*)(&Blds[srow][sk + 8]) = c1;
        }
        __syncthreads();

        #pragma unroll
        for (int kk = 0; kk < BK; kk += 32) {
            // A fragments: 2 M-subtiles, rows wave*32 + sm*16 + l16
            v16bf af[2];
            #pragma unroll
            for (int sm = 0; sm < 2; ++sm) {
                const __bf16* base = &Alds[wave * 32 + sm * 16 + l16][kk + half * 8];
                v8bf lo = *(const v8bf*)(base);
                v8bf hi = *(const v8bf*)(base + 16);
                #pragma unroll
                for (int i = 0; i < 8; ++i) { af[sm][i] = lo[i]; af[sm][i + 8] = hi[i]; }
            }
            // 4 B fragments, each used by both M-subtiles
            #pragma unroll
            for (int sn = 0; sn < 4; ++sn) {
                v16bf bfrag;
                const __bf16* base = &Blds[sn * 16 + l16][kk + half * 8];
                v8bf lo = *(const v8bf*)(base);
                v8bf hi = *(const v8bf*)(base + 16);
                #pragma unroll
                for (int i = 0; i < 8; ++i) { bfrag[i] = lo[i]; bfrag[i + 8] = hi[i]; }
                #pragma unroll
                for (int sm = 0; sm < 2; ++sm) {
                    acc[sm][sn] = __builtin_amdgcn_wmma_f32_16x16x32_bf16(
                        false, af[sm], false, bfrag, (short)0, acc[sm][sn], false, false);
                }
            }
        }
        __syncthreads();
    }

    // ---- epilogue ----
    const int seg = n0 >> 10;  // 0=Q 1=K 2=V 3=gate 4=alpha

    if (seg == 1) {
        // per-head (64-col) L2 norm; rows r / r+8 live in separate 16-lane halves
        #pragma unroll
        for (int sm = 0; sm < 2; ++sm) {
            #pragma unroll
            for (int r = 0; r < 8; ++r) {
                float ss = acc[sm][0][r] * acc[sm][0][r] + acc[sm][1][r] * acc[sm][1][r]
                         + acc[sm][2][r] * acc[sm][2][r] + acc[sm][3][r] * acc[sm][3][r];
                ss += __shfl_xor(ss, 1, 32);
                ss += __shfl_xor(ss, 2, 32);
                ss += __shfl_xor(ss, 4, 32);
                ss += __shfl_xor(ss, 8, 32);
                const float inv = 1.0f / fmaxf(sqrtf(ss), 1e-12f);
                #pragma unroll
                for (int sn = 0; sn < 4; ++sn) acc[sm][sn][r] *= inv;
            }
        }
    }

    const long baseSeg[5] = {0L, 33554432L, 67108864L, 100663296L, 134217728L};
    const long obase    = baseSeg[seg];
    const int  segWidth = (seg == 4) ? 16 : 1024;

    #pragma unroll
    for (int sm = 0; sm < 2; ++sm) {
        #pragma unroll
        for (int sn = 0; sn < 4; ++sn) {
            const int colg = n0 + sn * 16 + l16;
            if (colg < N_TOTAL) {
                const int cseg = colg - (seg << 10);
                #pragma unroll
                for (int r = 0; r < 8; ++r) {
                    const long m = m0 + wave * 32 + sm * 16 + r + 8 * half;
                    float v = acc[sm][sn][r];
                    if (seg == 3)      v = v / (1.0f + __expf(-v));    // SiLU
                    else if (seg == 4) v = 1.0f / (1.0f + __expf(-v)); // sigmoid
                    out[obase + m * segWidth + cseg] = v;
                }
            }
        }
    }
}

extern "C" void kernel_launch(void* const* d_in, const int* in_sizes, int n_in,
                              void* d_out, int out_size, void* d_ws, size_t ws_size,
                              hipStream_t stream) {
    const float* x = (const float*)d_in[0];   // (4, 8192, 1024) f32
    const float* W = (const float*)d_in[1];   // (4112, 1024)    f32
    float* out = (float*)d_out;

    dim3 grid((N_TOTAL + BN - 1) / BN,        // 65
              (int)(M_TOTAL / BM));           // 128
    fused_proj_wmma_kernel<<<grid, 256, 0, stream>>>(x, W, out);
}
